// TransformerMIL_SB_18923625906698
// MI455X (gfx1250) — matrix operand order repaired and checked by reading, not compiled, not run on hardware
//
#include <hip/hip_runtime.h>
#include <hip/hip_bf16.h>
#include <math.h>

// ---------------------------------------------------------------------------
// Types for CDNA5 WMMA (wave32, 16x16x32 bf16 -> f32)
// ---------------------------------------------------------------------------
typedef __attribute__((ext_vector_type(16))) __bf16 v16bf;
typedef __attribute__((ext_vector_type(8)))  float  v8f;

union FragBF { unsigned u[8]; v16bf v; };

__device__ __forceinline__ unsigned short f2bf(float x) {
    return (unsigned short)(__float_as_uint(x) >> 16);
}
__device__ __forceinline__ unsigned pk2(float lo, float hi) {
    return (__float_as_uint(lo) >> 16) | (__float_as_uint(hi) & 0xffff0000u);
}

#if __has_builtin(__builtin_amdgcn_global_load_async_to_lds_b128) && \
    __has_builtin(__builtin_amdgcn_s_wait_asynccnt)
#define HAVE_ASYNC_LDS 1
// Builtin prototype (from hipcc diagnostic): param0 is addrspace(1) v4i*.
typedef int v4i_b __attribute__((vector_size(16)));
#define GAS1 __attribute__((address_space(1)))
#define LAS3 __attribute__((address_space(3)))
#else
#define HAVE_ASYNC_LDS 0
#endif

// ---------------------------------------------------------------------------
// bf16-WMMA GEMM:  C[M,N] = act(A[M,K] @ B[K,N] + bias) (+ resid)
//   Block 256 threads (8 waves), tile 128x128, K-step 32.
//   Wave grid 4(M)x2(N); wave tile 32x64 -> 8 WMMAs per K-step.
//   Software pipeline: global loads for tile k+1 staged in registers while
//   WMMAs of tile k run; double-buffered LDS; one barrier per K-step.
//   ABF (template): A operand is bf16; OBF (template): bf16 output.
//   remap==1: output row m -> (m>>8)*257 + (m&255) + 1   (proj -> CLS-padded x)
//   act: 0 none, 1 exact GELU, 2 tanh, 3 sigmoid
//   All launched shapes are exact multiples of 128x128x32 (no tail guards).
// ---------------------------------------------------------------------------
#define GT 256
template <int ABF, int OBF>
__global__ __launch_bounds__(GT)
void k_gemm(const void* __restrict__ A, const float* __restrict__ Bw,
            const float* __restrict__ bias, const float* __restrict__ resid,
            void* __restrict__ C, int Kd,
            long lda, int ldb, int ldc, int act, int remap)
{
    __shared__ unsigned short Asm[2][128][36];   // [buf][m][k]
    __shared__ unsigned short Bsm[2][128][36];   // [buf][n][k] (transposed)

    const float*          Af  = (const float*)A;
    const unsigned short* A16 = (const unsigned short*)A;

    const int tid  = threadIdx.x;
    const int w    = tid >> 5;
    const int lane = tid & 31;
    const int h    = lane >> 4;
    const int l15  = lane & 15;
    const int wm   = w >> 1;           // 0..3 -> M offset 32*wm
    const int wn   = w & 1;            // 0..1 -> N offset 64*wn
    const int bx   = blockIdx.x, by = blockIdx.y;

    // per-thread staging coordinates (add-only addressing in the K loop)
    const int rA = tid >> 3;                 // A row base (0..31), +32*i
    const int cA = (tid & 7) << 2;           // A col (0..28)
    const int kB = (tid >> 7) << 2;          // B k base (0 or 4), +8*i
    const int nB = tid & 127;                // B col
    const float*          apF = Af  + (size_t)(by * 128 + rA) * (size_t)lda + cA;
    const unsigned short* apH = A16 + (size_t)(by * 128 + rA) * (size_t)lda + cA;
    const float*          bp0 = Bw  + (size_t)kB * (size_t)ldb + (size_t)bx * 128 + nB;

    v8f acc[2][4] = {};

    // register staging for the next tile
    uint2  aH[4];
    float4 aF[4];
    float  bF[4][4];

    auto loadTile = [&](int kt) {
        const int ko = kt * 32;
        #pragma unroll
        for (int i = 0; i < 4; ++i) {
            if (ABF) aH[i] = *(const uint2*) (apH + (size_t)(32 * i) * (size_t)lda + ko);
            else     aF[i] = *(const float4*)(apF + (size_t)(32 * i) * (size_t)lda + ko);
        }
        #pragma unroll
        for (int i = 0; i < 4; ++i) {
            const float* bp = bp0 + (size_t)(ko + 8 * i) * (size_t)ldb;
            bF[i][0] = bp[0];
            bF[i][1] = bp[ldb];
            bF[i][2] = bp[2 * (size_t)ldb];
            bF[i][3] = bp[3 * (size_t)ldb];
        }
    };
    auto storeTile = [&](int buf) {
        #pragma unroll
        for (int i = 0; i < 4; ++i) {
            uint2 p;
            if (ABF) p = aH[i];
            else { p.x = pk2(aF[i].x, aF[i].y); p.y = pk2(aF[i].z, aF[i].w); }
            *(uint2*)&Asm[buf][rA + 32 * i][cA] = p;
        }
        #pragma unroll
        for (int i = 0; i < 4; ++i) {
            uint2 p; p.x = pk2(bF[i][0], bF[i][1]); p.y = pk2(bF[i][2], bF[i][3]);
            *(uint2*)&Bsm[buf][nB][kB + 8 * i] = p;
        }
    };

    const int ksteps = Kd >> 5;
    loadTile(0);
    storeTile(0);
    __syncthreads();

    for (int kt = 0; kt < ksteps; ++kt) {
        const int  cur = kt & 1;
        const bool hn  = (kt + 1 < ksteps);
        if (hn) loadTile(kt + 1);                       // loads fly under compute
        if (kt + 2 < ksteps) {                          // cache prefetch 2 ahead
            const int ko2 = (kt + 2) * 32;
            if (ABF) __builtin_prefetch(apH + ko2, 0, 1);
            else     __builtin_prefetch(apF + ko2, 0, 1);
            __builtin_prefetch(bp0 + (size_t)ko2 * (size_t)ldb, 0, 1);
        }

        // ---- compute tile kt from LDS buffer cur ----
        FragBF a[2];
        #pragma unroll
        for (int mt = 0; mt < 2; ++mt) {
            const unsigned short* ar = &Asm[cur][32 * wm + 16 * mt + l15][0];
            #pragma unroll
            for (int j = 0; j < 4; ++j) {
                a[mt].u[j]     = *(const unsigned*)(ar + 8 * h + 2 * j);
                a[mt].u[4 + j] = *(const unsigned*)(ar + 16 + 8 * h + 2 * j);
            }
        }
        #pragma unroll
        for (int nt = 0; nt < 4; ++nt) {
            FragBF b;
            const unsigned short* br = &Bsm[cur][64 * wn + 16 * nt + l15][0];
            #pragma unroll
            for (int v = 0; v < 8; ++v)
                b.u[v] = *(const unsigned*)(br + (v >> 2) * 16 + 8 * h + 2 * (v & 3));
            #pragma unroll
            for (int mt = 0; mt < 2; ++mt)
                acc[mt][nt] = __builtin_amdgcn_wmma_f32_16x16x32_bf16(
                    false, a[mt].v, false, b.v, (short)0, acc[mt][nt], false, false);
        }

        if (hn) storeTile(cur ^ 1);                     // fill the other buffer
        __syncthreads();
    }

    // epilogue (D layout: m = 8*h + r, n = l15)
    #pragma unroll
    for (int mt = 0; mt < 2; ++mt) {
        #pragma unroll
        for (int nt = 0; nt < 4; ++nt) {
            #pragma unroll
            for (int r = 0; r < 8; ++r) {
                const int gm = by * 128 + 32 * wm + 16 * mt + 8 * h + r;
                const int gn = bx * 128 + 64 * wn + 16 * nt + l15;
                float val = acc[mt][nt][r];
                if (bias) val += bias[gn];
                if (act == 1)      val = 0.5f * val * (1.0f + erff(val * 0.70710678f));
                else if (act == 2) val = tanhf(val);
                else if (act == 3) val = 1.0f / (1.0f + __expf(-val));
                const long row = remap ? ((long)(gm >> 8) * 257 + (gm & 255) + 1)
                                       : (long)gm;
                const size_t off = (size_t)row * (size_t)ldc + gn;
                if (OBF) {
                    ((unsigned short*)C)[off] = f2bf(val);
                } else {
                    if (resid) val += resid[off];
                    ((float*)C)[off] = val;
                }
            }
        }
    }
}

// ---------------------------------------------------------------------------
// CLS token fill: x[b, 0, :] = cls[:]
// ---------------------------------------------------------------------------
__global__ void k_fill_cls(const float* __restrict__ cls, float* __restrict__ x)
{
    const int i = blockIdx.x * blockDim.x + threadIdx.x;   // 256*512
    const int b = i >> 9, d = i & 511;
    x[(size_t)(b * 257) * 512 + d] = cls[d];
}

// ---------------------------------------------------------------------------
// LayerNorm over D=512, one block (128 threads) per row; bf16 output
// ---------------------------------------------------------------------------
__global__ __launch_bounds__(128)
void k_ln(const float* __restrict__ x, const float* __restrict__ g,
          const float* __restrict__ b, unsigned short* __restrict__ y)
{
    const int row = blockIdx.x;
    const int tid = threadIdx.x;
    const float4 v = *(const float4*)(x + (size_t)row * 512 + tid * 4);
    __shared__ float s1[128], s2[128];
    s1[tid] = v.x + v.y + v.z + v.w;
    s2[tid] = v.x * v.x + v.y * v.y + v.z * v.z + v.w * v.w;
    __syncthreads();
    for (int off = 64; off > 0; off >>= 1) {
        if (tid < off) { s1[tid] += s1[tid + off]; s2[tid] += s2[tid + off]; }
        __syncthreads();
    }
    const float mean = s1[0] * (1.0f / 512.0f);
    const float var  = s2[0] * (1.0f / 512.0f) - mean * mean;
    const float rs   = rsqrtf(var + 1e-5f);
    const float4 gg = *(const float4*)(g + tid * 4);
    const float4 bb = *(const float4*)(b + tid * 4);
    uint2 p;
    p.x = pk2((v.x - mean) * rs * gg.x + bb.x, (v.y - mean) * rs * gg.y + bb.y);
    p.y = pk2((v.z - mean) * rs * gg.z + bb.z, (v.w - mean) * rs * gg.w + bb.w);
    *(uint2*)(y + (size_t)row * 512 + tid * 4) = p;
}

// ---------------------------------------------------------------------------
// Fused attention, one block (4 waves) per (batch, head); bf16 in, bf16 out.
// Tokens 257 padded to 272 (scores) / 288 (PV K-dim). DH = 64.
// qkv layout: bf16 [b*257+t][1536], q|k|v at +0|+512|+1024, head slice hh*64.
// ---------------------------------------------------------------------------
#define AW 4
#define ATTN_LDS ((272*64 + 64*288 + AW*16*64 + AW*16*288) * 2)
__global__ __launch_bounds__(128)
void k_attn(const unsigned short* __restrict__ qkv, unsigned short* __restrict__ o)
{
    extern __shared__ char smem[];
    unsigned short* Kmat = (unsigned short*)smem;       // [272][64]   key-major
    unsigned short* Vt   = Kmat + 272 * 64;             // [64][288]   dh-major (V^T)
    unsigned short* Qs   = Vt + 64 * 288;               // [AW][16][64]
    unsigned short* Ps   = Qs + AW * 16 * 64;           // [AW][16][288]

    const int tid  = threadIdx.x;
    const int w    = tid >> 5;
    const int lane = tid & 31;
    const int h    = lane >> 4;
    const int l15  = lane & 15;
    const int bh   = blockIdx.x;
    const int bb   = bh >> 3, hh = bh & 7;
    const size_t rowbase = (size_t)bb * 257;

    // ---- stage K rows ----
#if HAVE_ASYNC_LDS
    for (int idx = tid; idx < 257 * 8; idx += 128) {     // 16B chunks, 8 per row
        const int row = idx >> 3;
        const int ch  = (idx & 7) << 3;                  // bf16 offset in row
        const unsigned short* gp = qkv + (rowbase + row) * 1536 + 512 + hh * 64 + ch;
        __builtin_amdgcn_global_load_async_to_lds_b128(
            (GAS1 v4i_b*)(const void*)gp,
            (LAS3 v4i_b*)(void*)&Kmat[row * 64 + ch], 0, 0);
    }
    for (int idx = tid; idx < 15 * 16; idx += 128) {     // zero pad rows 257..271
        const int row = 257 + (idx >> 4);
        const int c4  = (idx & 15) << 2;
        uint2 z; z.x = 0u; z.y = 0u;
        *(uint2*)&Kmat[row * 64 + c4] = z;
    }
#else
    for (int idx = tid; idx < 272 * 16; idx += 128) {
        const int row = idx >> 4;
        const int c4  = (idx & 15) << 2;
        uint2 kp; kp.x = 0u; kp.y = 0u;
        if (row < 257)
            kp = *(const uint2*)(qkv + (rowbase + row) * 1536 + 512 + hh * 64 + c4);
        *(uint2*)&Kmat[row * 64 + c4] = kp;
    }
#endif
    // ---- stage V^T (element scatter) ----
    for (int idx = tid; idx < 288 * 16; idx += 128) {
        const int row = idx >> 4;             // key 0..287
        const int c4  = (idx & 15) << 2;      // dh 0..60
        unsigned short e0 = 0, e1 = 0, e2 = 0, e3 = 0;
        if (row < 257) {
            const uint2 vp = *(const uint2*)(qkv + (rowbase + row) * 1536 + 1024 + hh * 64 + c4);
            e0 = (unsigned short)(vp.x & 0xffffu);
            e1 = (unsigned short)(vp.x >> 16);
            e2 = (unsigned short)(vp.y & 0xffffu);
            e3 = (unsigned short)(vp.y >> 16);
        }
        Vt[(c4 + 0) * 288 + row] = e0;
        Vt[(c4 + 1) * 288 + row] = e1;
        Vt[(c4 + 2) * 288 + row] = e2;
        Vt[(c4 + 3) * 288 + row] = e3;
    }
    // zero P pad columns [272, 288)
    for (int idx = tid; idx < AW * 16 * 16; idx += 128) {
        const int wr = idx >> 8;
        const int r  = (idx >> 4) & 15;
        const int c  = 272 + (idx & 15);
        Ps[(wr * 16 + r) * 288 + c] = 0;
    }
#if HAVE_ASYNC_LDS
    __builtin_amdgcn_s_wait_asynccnt(0);
#endif
    __syncthreads();

    // 17 query tiles / 4 waves -> uniform 5 iterations (non-divergent barriers)
    for (int it = 0; it < 5; ++it) {
        const int qt = w + AW * it;
        const bool active = (qt < 17);

        if (active) {   // stage this wave's Q tile (bf16 copy)
            for (int idx = lane; idx < 16 * 8; idx += 32) {
                const int r = idx >> 3, ch = (idx & 7) << 3;
                const int token = qt * 16 + r;
                uint4 qv; qv.x = qv.y = qv.z = qv.w = 0u;
                if (token < 257)
                    qv = *(const uint4*)(qkv + (rowbase + token) * 1536 + hh * 64 + ch);
                *(uint4*)(Qs + (w * 16 + r) * 64 + ch) = qv;
            }
        }
        __syncthreads();

        if (active) {
            // scores: S = Q @ K^T, 17 n-tiles, K-dim 64 (2 WMMA k-steps)
            v8f sc[17] = {};
            const unsigned short* qr = Qs + (w * 16 + l15) * 64;
            #pragma unroll
            for (int ks = 0; ks < 2; ++ks) {
                FragBF a;
                #pragma unroll
                for (int j = 0; j < 4; ++j) {
                    a.u[j]     = *(const unsigned*)(qr + ks * 32 + 8 * h + 2 * j);
                    a.u[4 + j] = *(const unsigned*)(qr + ks * 32 + 16 + 8 * h + 2 * j);
                }
                #pragma unroll
                for (int nt = 0; nt < 17; ++nt) {
                    FragBF b;
                    const unsigned short* br = Kmat + (nt * 16 + l15) * 64;
                    #pragma unroll
                    for (int v = 0; v < 8; ++v)
                        b.u[v] = *(const unsigned*)(br + ks * 32 + (v >> 2) * 16 + 8 * h + 2 * (v & 3));
                    sc[nt] = __builtin_amdgcn_wmma_f32_16x16x32_bf16(
                        false, a.v, false, b.v, (short)0, sc[nt], false, false);
                }
            }
            // row softmax (row m = 8h + r lives in reg r across 16 lanes of half h)
            #pragma unroll
            for (int r = 0; r < 8; ++r) {
                float mx = -1e30f;
                #pragma unroll
                for (int nt = 0; nt < 17; ++nt) {
                    float s = sc[nt][r] * 0.125f;            // DH^-0.5
                    if (nt == 16 && l15 != 0) s = -1e30f;    // mask cols >= 257
                    sc[nt][r] = s;
                    mx = fmaxf(mx, s);
                }
                #pragma unroll
                for (int m = 1; m < 16; m <<= 1)
                    mx = fmaxf(mx, __shfl_xor(mx, m, 32));
                float sum = 0.f;
                #pragma unroll
                for (int nt = 0; nt < 17; ++nt) {
                    const float e = __expf(sc[nt][r] - mx);
                    sc[nt][r] = e;
                    sum += e;
                }
                #pragma unroll
                for (int m = 1; m < 16; m <<= 1)
                    sum += __shfl_xor(sum, m, 32);
                const float inv = 1.0f / sum;
                unsigned short* prow = Ps + (w * 16 + 8 * h + r) * 288;
                #pragma unroll
                for (int nt = 0; nt < 17; ++nt)
                    prow[nt * 16 + l15] = f2bf(sc[nt][r] * inv);
            }
        }
        __syncthreads();

        if (active) {
            // O = P @ V   (K-dim 288 -> 9 WMMA k-steps, 4 dh n-tiles)
            v8f oc[4] = {};
            const unsigned short* pr = Ps + (w * 16 + l15) * 288;
            #pragma unroll
            for (int ks = 0; ks < 9; ++ks) {
                FragBF a;
                #pragma unroll
                for (int j = 0; j < 4; ++j) {
                    a.u[j]     = *(const unsigned*)(pr + ks * 32 + 8 * h + 2 * j);
                    a.u[4 + j] = *(const unsigned*)(pr + ks * 32 + 16 + 8 * h + 2 * j);
                }
                #pragma unroll
                for (int nt = 0; nt < 4; ++nt) {
                    FragBF b;
                    const unsigned short* br = Vt + (nt * 16 + l15) * 288;
                    #pragma unroll
                    for (int v = 0; v < 8; ++v)
                        b.u[v] = *(const unsigned*)(br + ks * 32 + (v >> 2) * 16 + 8 * h + 2 * (v & 3));
                    oc[nt] = __builtin_amdgcn_wmma_f32_16x16x32_bf16(
                        false, a.v, false, b.v, (short)0, oc[nt], false, false);
                }
            }
            #pragma unroll
            for (int nt = 0; nt < 4; ++nt) {
                #pragma unroll
                for (int r = 0; r < 8; ++r) {
                    const int token = qt * 16 + 8 * h + r;
                    if (token < 257)
                        o[(rowbase + token) * 512 + hh * 64 + nt * 16 + l15] = f2bf(oc[nt][r]);
                }
            }
        }
        __syncthreads();
    }
}

// ---------------------------------------------------------------------------
// Attention pooling gate: Araw[b] = sum_d AV[b,d]*AU[b,d]*aw[d] + awb
// ---------------------------------------------------------------------------
__global__ __launch_bounds__(128)
void k_attnpool(const float* __restrict__ AV, const float* __restrict__ AU,
                const float* __restrict__ aw, const float* __restrict__ awb,
                float* __restrict__ Araw)
{
    const int b = blockIdx.x, tid = threadIdx.x;
    float s = 0.f;
    for (int d = tid; d < 512; d += 128)
        s += AV[b * 512 + d] * AU[b * 512 + d] * aw[d];
    __shared__ float red[128];
    red[tid] = s;
    __syncthreads();
    for (int off = 64; off > 0; off >>= 1) {
        if (tid < off) red[tid] += red[tid + off];
        __syncthreads();
    }
    if (tid == 0) Araw[b] = red[0] + awb[0];
}

// ---------------------------------------------------------------------------
// Head: softmax over 256, top-K selection, instance loss, bag classifier.
// Output (294 floats): logits[2], Y_prob[2], Y_hat[1], A_raw[256],
//                      loss[1], inst_preds[16], inst_tgts[16]
// ---------------------------------------------------------------------------
__global__ __launch_bounds__(256)
void k_head(const float* __restrict__ Araw, const float* __restrict__ x,
            const float* __restrict__ cls_w, const float* __restrict__ cls_b,
            const float* __restrict__ inst_w, const float* __restrict__ inst_b,
            const int* __restrict__ label, float* __restrict__ out)
{
    const int t = threadIdx.x;
    __shared__ float sA[256], sS[256], Msh[512], red[256], sLoss[16];
    __shared__ int sel[16];

    sA[t] = Araw[t];
    out[5 + t] = sA[t];                       // A_raw
    __syncthreads();

    // softmax over 256
    red[t] = sA[t]; __syncthreads();
    for (int off = 128; off > 0; off >>= 1) {
        if (t < off) red[t] = fmaxf(red[t], red[t + off]);
        __syncthreads();
    }
    const float mx = red[0]; __syncthreads();
    const float e = __expf(sA[t] - mx);
    red[t] = e; sS[t] = e; __syncthreads();
    for (int off = 128; off > 0; off >>= 1) {
        if (t < off) red[t] += red[t + off];
        __syncthreads();
    }
    const float inv = 1.0f / red[0]; __syncthreads();
    sS[t] *= inv;                              // A_s
    __syncthreads();

    // M = A_s @ Hc
    for (int d = t; d < 512; d += 256) {
        float acc = 0.f;
        for (int b = 0; b < 256; ++b)
            acc += sS[b] * x[(size_t)(b * 257) * 512 + d];
        Msh[d] = acc;
    }
    __syncthreads();

    if (t < 2) {
        float lg = cls_b[t];
        for (int d = 0; d < 512; ++d) lg += Msh[d] * cls_w[d * 2 + t];
        out[t] = lg;
        red[t] = lg;
    }
    __syncthreads();
    if (t == 0) {
        const float l0 = red[0], l1 = red[1];
        const float m = fmaxf(l0, l1);
        const float z = __expf(l0 - m) + __expf(l1 - m);
        out[2] = __expf(l0 - m) / z;
        out[3] = __expf(l1 - m) / z;
        out[4] = (l1 > l0) ? 1.0f : 0.0f;
    }
    __syncthreads();

    // top-8 largest
    sA[t] = sS[t]; __syncthreads();
    if (t == 0) {
        for (int i = 0; i < 8; ++i) {
            int bi = 0; float bv = -1e30f;
            for (int j = 0; j < 256; ++j)
                if (sA[j] > bv) { bv = sA[j]; bi = j; }
            sel[i] = bi; sA[bi] = -1e30f;
        }
    }
    __syncthreads();
    // top-8 smallest
    sA[t] = sS[t]; __syncthreads();
    if (t == 0) {
        for (int i = 0; i < 8; ++i) {
            int bi = 0; float bv = 1e30f;
            for (int j = 0; j < 256; ++j)
                if (sA[j] < bv) { bv = sA[j]; bi = j; }
            sel[8 + i] = bi; sA[bi] = 1e30f;
        }
    }
    __syncthreads();

    const int lab = label[0];
    const float* ww = inst_w + (size_t)lab * 512 * 2;
    const float* ib = inst_b + lab * 2;
    if (t < 16) {
        const float* hrow = x + (size_t)(sel[t] * 257) * 512;
        float lg0 = ib[0], lg1 = ib[1];
        for (int d = 0; d < 512; ++d) {
            const float hv = hrow[d];
            lg0 += hv * ww[d * 2 + 0];
            lg1 += hv * ww[d * 2 + 1];
        }
        const int tgt = (t < 8) ? 1 : 0;
        const float m = fmaxf(lg0, lg1);
        const float lse = m + __logf(__expf(lg0 - m) + __expf(lg1 - m));
        const float lp = (tgt ? lg1 : lg0) - lse;
        sLoss[t] = -lp;
        out[262 + t] = (lg1 > lg0) ? 1.0f : 0.0f;
        out[278 + t] = (float)tgt;
    }
    __syncthreads();
    if (t == 0) {
        float L = 0.f;
        for (int i = 0; i < 16; ++i) L += sLoss[i];
        out[261] = L * (1.0f / 16.0f);
    }
}

// ---------------------------------------------------------------------------
// Host driver
// ---------------------------------------------------------------------------
extern "C" void kernel_launch(void* const* d_in, const int* in_sizes, int n_in,
                              void* d_out, int out_size, void* d_ws, size_t ws_size,
                              hipStream_t stream)
{
    (void)in_sizes; (void)n_in; (void)out_size; (void)ws_size;

    // Inputs in setup_inputs() insertion order: xs, label, params{...}
    const float* xs     = (const float*)d_in[0];
    const int*   label  = (const int*)  d_in[1];
    const float* proj_w = (const float*)d_in[2];
    const float* proj_b = (const float*)d_in[3];
    const float* clstok = (const float*)d_in[4];
    struct Layer { const float *ln1_g, *ln1_b, *wqkv, *wo, *bo, *ln2_g, *ln2_b, *w1, *b1, *w2, *b2; };
    Layer L[2];
    for (int i = 0; i < 2; ++i) {
        const int base = 5 + i * 11;
        L[i].ln1_g = (const float*)d_in[base + 0];
        L[i].ln1_b = (const float*)d_in[base + 1];
        L[i].wqkv  = (const float*)d_in[base + 2];
        L[i].wo    = (const float*)d_in[base + 3];
        L[i].bo    = (const float*)d_in[base + 4];
        L[i].ln2_g = (const float*)d_in[base + 5];
        L[i].ln2_b = (const float*)d_in[base + 6];
        L[i].w1    = (const float*)d_in[base + 7];
        L[i].b1    = (const float*)d_in[base + 8];
        L[i].w2    = (const float*)d_in[base + 9];
        L[i].b2    = (const float*)d_in[base + 10];
    }
    const float* v2_w   = (const float*)d_in[27];
    const float* v2_b   = (const float*)d_in[28];
    const float* u2_w   = (const float*)d_in[29];
    const float* u2_b   = (const float*)d_in[30];
    const float* aw2_w  = (const float*)d_in[31];
    const float* aw2_b  = (const float*)d_in[32];
    const float* cls_w  = (const float*)d_in[33];
    const float* cls_b  = (const float*)d_in[34];
    const float* inst_w = (const float*)d_in[35];
    const float* inst_b = (const float*)d_in[36];

    const int ROWS = 256 * 257;   // 65792 tokens incl. CLS

    // workspace carve
    char* ws = (char*)d_ws;
    float*          x     = (float*)ws;           ws += (size_t)ROWS * 512 * 4;
    unsigned short* hbf   = (unsigned short*)ws;  ws += (size_t)ROWS * 512 * 2;
    unsigned short* obf   = (unsigned short*)ws;  ws += (size_t)ROWS * 512 * 2;
    unsigned short* big16 = (unsigned short*)ws;  ws += (size_t)ROWS * 2048 * 2;  // qkv / mlp hidden
    float*          AV    = (float*)ws;           ws += (size_t)256 * 512 * 4;
    float*          AU    = (float*)ws;           ws += (size_t)256 * 512 * 4;
    float*          Araw  = (float*)ws;           ws += 256 * 4;

    const dim3 gblk(GT);

    // input projection (remapped into CLS-padded x) + CLS fill
    k_gemm<0, 0><<<dim3(512 / 128, 65536 / 128), gblk, 0, stream>>>(
        xs, proj_w, proj_b, nullptr, x, 1024, 1024, 512, 512, 0, 1);
    k_fill_cls<<<(256 * 512) / 256, 256, 0, stream>>>(clstok, x);

    for (int i = 0; i < 2; ++i) {
        k_ln<<<ROWS, 128, 0, stream>>>(x, L[i].ln1_g, L[i].ln1_b, hbf);
        k_gemm<1, 1><<<dim3(1536 / 128, ROWS / 128), gblk, 0, stream>>>(
            hbf, L[i].wqkv, nullptr, nullptr, big16, 512, 512, 1536, 1536, 0, 0);
        k_attn<<<256 * 8, 128, ATTN_LDS, stream>>>(big16, obf);
        k_gemm<1, 0><<<dim3(512 / 128, ROWS / 128), gblk, 0, stream>>>(
            obf, L[i].wo, L[i].bo, x, x, 512, 512, 512, 512, 0, 0);
        k_ln<<<ROWS, 128, 0, stream>>>(x, L[i].ln2_g, L[i].ln2_b, hbf);
        k_gemm<1, 1><<<dim3(2048 / 128, ROWS / 128), gblk, 0, stream>>>(
            hbf, L[i].w1, L[i].b1, nullptr, big16, 512, 512, 2048, 2048, 1, 0);
        k_gemm<1, 0><<<dim3(512 / 128, ROWS / 128), gblk, 0, stream>>>(
            big16, L[i].w2, L[i].b2, x, x, 2048, 2048, 512, 512, 0, 0);
    }

    // gated attention pooling head (Hc = x[:,0,:] via lda = 257*512)
    k_gemm<0, 0><<<dim3(512 / 128, 256 / 128), gblk, 0, stream>>>(
        x, v2_w, v2_b, nullptr, AV, 512, 257L * 512, 512, 512, 2, 0);
    k_gemm<0, 0><<<dim3(512 / 128, 256 / 128), gblk, 0, stream>>>(
        x, u2_w, u2_b, nullptr, AU, 512, 257L * 512, 512, 512, 3, 0);
    k_attnpool<<<256, 128, 0, stream>>>(AV, AU, aw2_w, aw2_b, Araw);
    k_head<<<1, 256, 0, stream>>>(Araw, x, cls_w, cls_b, inst_w, inst_b, label,
                                  (float*)d_out);
}
